// ED_DCNN_27573690040389
// MI455X (gfx1250) — matrix operand order repaired and checked by reading, not compiled, not run on hardware
//
#include <hip/hip_runtime.h>
#include <hip/hip_bf16.h>

typedef __bf16 bf16;
typedef __attribute__((ext_vector_type(16))) __bf16 v16bf;
typedef __attribute__((ext_vector_type(8)))  float  v8f;

// ---------------------------------------------------------------------------
// Weight prep: fp32 (Cout,Cin,3,3) -> bf16, blocked + K-order swizzled:
//   dst[((octile*(Cin/32) + chunk)*9 + kk)*2048 + m*32 + pos]
//   octile=oc/64, m=oc%64, chunk=c/32, pos = c%32
// so the conv kernel can stage one 18432-element block with raw 16B copies.
// ---------------------------------------------------------------------------
__global__ __launch_bounds__(256)
void k_prep_w(const float* __restrict__ w, bf16* __restrict__ out,
              int Cout, int Cin, int total) {
    int idx = blockIdx.x * 256 + threadIdx.x;
    if (idx >= total) return;
    int kk = idx % 9;
    int t  = idx / 9;
    int c  = t % Cin;
    int oc = t / Cin;
    int octile = oc >> 6, m = oc & 63;
    int chunk  = c >> 5,  cl = c & 31;
    int blk = cl >> 3;
    int pos = blk * 8 + (cl & 7);
    out[((size_t)(octile * (Cin >> 5) + chunk) * 9 + kk) * 2048 + m * 32 + pos] =
        (bf16)w[idx];
}

// ---------------------------------------------------------------------------
// x (B=4,C=256,T=8,H=64,W=64) fp32  ->  nd (N=32,C=256,64,64) bf16,  n = b*8+t
// ---------------------------------------------------------------------------
__global__ __launch_bounds__(256) void k_x_to_nd(const float* __restrict__ x,
                                                  bf16* __restrict__ out, int total) {
    int idx = blockIdx.x * 256 + threadIdx.x;
    if (idx >= total) return;
    int p = idx & 4095;
    int c = (idx >> 12) & 255;
    int n = idx >> 20;
    int b = n >> 3, t = n & 7;
    out[idx] = (bf16)x[(((size_t)(b * 256 + c)) * 8 + t) * 4096 + p];
}

// ---------------------------------------------------------------------------
// 3x3 SAME conv via implicit GEMM + bf16 WMMA, register-tiled.
//   in  : bf16 (N, Cin, 64, 64)
//   wgt : pre-swizzled bf16 blocks from k_prep_w
// Block: 256 thr = 8 waves, wave tile = 32 oc x 64 px (2 A-frags x 4 B-frags,
// 8 accumulators), waves arranged 2(M) x 4(N) -> block tile 64 oc x 4 rows.
// Grid: (16, Cout/64, 32).
//   mode 0: store f32 (n,Cout,HW)                              (offset convs)
//   mode 1: BN+LeakyReLU, store f32 + bf16 (n,Cout,HW)         (conv1)
//   mode 2: BN+LeakyReLU, store f32 transposed into d_out      (conv2)
// ---------------------------------------------------------------------------
__global__ __launch_bounds__(256)
void k_conv3x3_wmma(const bf16* __restrict__ in, const bf16* __restrict__ wgt,
                    int Cin,
                    float* __restrict__ outf, bf16* __restrict__ outbf,
                    const float* __restrict__ gg, const float* __restrict__ bb,
                    const float* __restrict__ mm, const float* __restrict__ vv,
                    int mode) {
    __shared__ __align__(64) bf16 lds_in[6 * 66 * 32];   // [row 6][col 66][c 32]
    __shared__ __align__(64) bf16 lds_w[9 * 64 * 32];    // [tap][oc][K-ordered c]

    const int tid    = threadIdx.x;
    const int n      = blockIdx.z;
    const int ocbase = blockIdx.y * 64;
    const int Cout   = gridDim.y * 64;
    const int y0     = blockIdx.x * 4;      // 4 output rows per block

    const int wv   = tid >> 5;
    const int lane = tid & 31;
    const int wM   = wv & 1;     // out-channel half (32 oc)
    const int wN   = wv >> 1;    // output row within the 4-row tile
    const int half = lane >> 4;
    const int ln   = lane & 15;

    const bf16* __restrict__ inN = in + (size_t)n * Cin * 4096;
    const int nchunk = Cin >> 5;

    // halo columns x=-1 (col 0) and x=64 (col 65) are identically zero for all
    // K-chunks: write them once, outside the chunk loop.
    for (int idx = tid; idx < 6 * 2 * 32; idx += 256) {
        int c   = idx & 31;
        int r   = idx >> 5;         // 0..11
        int row = r >> 1;
        int col = (r & 1) ? 65 : 0;
        lds_in[(row * 66 + col) * 32 + c] = (bf16)0.0f;
    }

    v8f acc[2][4] = {};

    for (int ck = 0; ck < nchunk; ++ck) {
        const int c0 = ck << 5;
        __syncthreads();   // protect previous iteration's LDS reads (and edge init)
        // ---- stage input interior: rows y0-1..y0+4, x 0..63 (cols 1..64) ----
        for (int idx = tid; idx < 6 * 64 * 32; idx += 256) {
            int c   = idx / 384;
            int r   = idx - c * 384;
            int row = r >> 6;
            int x   = r & 63;
            int y = y0 - 1 + row;
            bf16 val = (bf16)0.0f;
            if (y >= 0 && y < 64)
                val = inN[(size_t)(c0 + c) * 4096 + y * 64 + x];
            lds_in[(row * 66 + x + 1) * 32 + c] = val;
        }
        // ---- stage weights: one pre-swizzled 18432-elem block, 16B copies ----
        {
            const bf16* wsrc = wgt + (size_t)((ocbase >> 6) * nchunk + ck) * 18432;
            for (int i = tid; i < 18432 / 8; i += 256)
                *(uint4*)(&lds_w[i * 8]) = *(const uint4*)(&wsrc[i * 8]);
        }
        __syncthreads();

        // prefetch next K-chunk while WMMAs run
        if (c0 + 32 < Cin) {
            __builtin_prefetch(&inN[(size_t)(c0 + 32 + (tid & 31)) * 4096 + y0 * 64], 0, 1);
            __builtin_prefetch(&wgt[(size_t)((ocbase >> 6) * nchunk + ck + 1) * 18432 +
                                    tid * 72], 0, 1);
        }

        // ---- 9 taps x (2 A-frags x 4 B-frags) WMMAs ----
#pragma unroll
        for (int kk = 0; kk < 9; ++kk) {
            const int ky = kk / 3, kx = kk - ky * 3;
            v16bf a0 = *(const v16bf*)&lds_w[kk * 2048 + (wM * 32 + ln) * 32 + 16 * half];
            v16bf a1 = *(const v16bf*)&lds_w[kk * 2048 + (wM * 32 + 16 + ln) * 32 + 16 * half];
#pragma unroll
            for (int ni = 0; ni < 4; ++ni) {
                const bf16* bp =
                    &lds_in[((wN + ky) * 66 + kx + 16 * ni + ln) * 32 + 16 * half];
                v16bf b = *(const v16bf*)bp;
                acc[0][ni] = __builtin_amdgcn_wmma_f32_16x16x32_bf16(
                    false, a0, false, b, (short)0, acc[0][ni], false, false);
                acc[1][ni] = __builtin_amdgcn_wmma_f32_16x16x32_bf16(
                    false, a1, false, b, (short)0, acc[1][ni], false, false);
            }
        }
    }

    // ---- store (C/D layout: VGPR i -> M = i + 8*half, N = ln) ----
    const int row = y0 + wN;
#pragma unroll
    for (int mi = 0; mi < 2; ++mi) {
#pragma unroll
        for (int ni = 0; ni < 4; ++ni) {
            const int pix = row * 64 + ni * 16 + ln;
#pragma unroll
            for (int i = 0; i < 8; ++i) {
                int ch = ocbase + wM * 32 + mi * 16 + 8 * half + i;
                float v = acc[mi][ni][i];
                if (mode != 0) {
                    float s = gg[ch] * rsqrtf(vv[ch] + 1e-5f);
                    v = (v - mm[ch]) * s + bb[ch];
                    v = v > 0.0f ? v : 0.01f * v;
                }
                if (mode == 2) {
                    int bi = n >> 3, ti = n & 7;
                    outf[(((size_t)(bi * Cout + ch)) * 8 + ti) * 4096 + pix] = v;
                } else {
                    size_t o = ((size_t)n * Cout + ch) * 4096 + pix;
                    outf[o] = v;
                    if (mode == 1) outbf[o] = (bf16)v;
                }
            }
        }
    }
}

// ---------------------------------------------------------------------------
// Deformable bilinear sampling (ConvOffset2D semantics, incl. the contiguous
// (N,2C,H,W)->(N*C,H,W,2) view: offset pair at base(2c) + h*2W + 2w).
//   srcmode 1: src is original x fp32 (B,C,T,H,W), n = b*8+t
//   srcmode 0: src is fp32 (N, C, H, W)
// out: bf16 (N, C, H, W)
// ---------------------------------------------------------------------------
__global__ __launch_bounds__(256)
void k_deform_sample(const float* __restrict__ src, int srcmode,
                     const float* __restrict__ off, int C,
                     bf16* __restrict__ out, int total) {
    int idx = blockIdx.x * 256 + threadIdx.x;
    if (idx >= total) return;
    int p = idx & 4095;
    int c = (idx >> 12) % C;
    int n = idx / (C << 12);
    int h = p >> 6, w = p & 63;

    const float* ob = off + ((size_t)(n * 2 * C + 2 * c) << 12);
    float oy = ob[h * 128 + 2 * w];
    float ox = ob[h * 128 + 2 * w + 1];

    float cy = fminf(fmaxf((float)h + oy, 0.0f), 63.0f);
    float cx = fminf(fmaxf((float)w + ox, 0.0f), 63.0f);
    float y0f = floorf(cy), x0f = floorf(cx);
    int   iy0 = (int)y0f,  ix0 = (int)x0f;
    float ty = cy - y0f,   tx = cx - x0f;
    int   iy1 = iy0 + 1 < 64 ? iy0 + 1 : 63;
    int   ix1 = ix0 + 1 < 64 ? ix0 + 1 : 63;

    const float* sb;
    if (srcmode == 1) {
        int b = n >> 3, t = n & 7;
        sb = src + (((size_t)(b * C + c)) * 8 + t) * 4096;
    } else {
        sb = src + ((size_t)(n * C + c)) * 4096;
    }
    float v00 = sb[iy0 * 64 + ix0], v01 = sb[iy0 * 64 + ix1];
    float v10 = sb[iy1 * 64 + ix0], v11 = sb[iy1 * 64 + ix1];
    float r = v00 * (1.f - ty) * (1.f - tx) + v01 * (1.f - ty) * tx +
              v10 * ty * (1.f - tx) + v11 * ty * tx;
    out[idx] = (bf16)r;
}

// ---------------------------------------------------------------------------
extern "C" void kernel_launch(void* const* d_in, const int* in_sizes, int n_in,
                              void* d_out, int out_size, void* d_ws, size_t ws_size,
                              hipStream_t stream) {
    const float* x      = (const float*)d_in[0];
    const float* w_off1 = (const float*)d_in[3];
    const float* w1     = (const float*)d_in[4];
    const float* g1 = (const float*)d_in[5],  *b1 = (const float*)d_in[6];
    const float* m1 = (const float*)d_in[7],  *v1 = (const float*)d_in[8];
    const float* w_off2 = (const float*)d_in[9];
    const float* w2     = (const float*)d_in[10];
    const float* g2 = (const float*)d_in[11], *b2 = (const float*)d_in[12];
    const float* m2 = (const float*)d_in[13], *v2 = (const float*)d_in[14];

    char* ws = (char*)d_ws;
    size_t o = 0;
    auto alloc = [&](size_t bytes) -> void* {
        void* p = ws + o;
        o += (bytes + 255) & ~(size_t)255;
        return p;
    };

    const int NIMG = 32, HW = 4096;
    const int nwoff1 = 512 * 256 * 9, nw1 = 64 * 256 * 9;
    const int nwoff2 = 128 * 64 * 9,  nw2 = 64 * 64 * 9;

    bf16* xnd     = (bf16*)alloc((size_t)NIMG * 256 * HW * 2);
    bf16* woff1b  = (bf16*)alloc((size_t)nwoff1 * 2);
    bf16* w1b     = (bf16*)alloc((size_t)nw1 * 2);
    bf16* woff2b  = (bf16*)alloc((size_t)nwoff2 * 2);
    bf16* w2b     = (bf16*)alloc((size_t)nw2 * 2);
    float* offbuf = (float*)alloc((size_t)NIMG * 512 * HW * 4);   // reused for off2
    bf16* sampb   = (bf16*)alloc((size_t)NIMG * 256 * HW * 2);    // reused stage 2
    float* midf   = (float*)alloc((size_t)NIMG * 64 * HW * 4);
    bf16* midb    = (bf16*)alloc((size_t)NIMG * 64 * HW * 2);
    (void)ws_size; (void)in_sizes; (void)n_in; (void)out_size;

    auto blk = [](int n) { return (n + 255) / 256; };

    // weight prep (fp32 -> bf16 + LDS-ready swizzle/blocking) + input convert
    k_prep_w<<<blk(nwoff1), 256, 0, stream>>>(w_off1, woff1b, 512, 256, nwoff1);
    k_prep_w<<<blk(nw1),    256, 0, stream>>>(w1,     w1b,     64, 256, nw1);
    k_prep_w<<<blk(nwoff2), 256, 0, stream>>>(w_off2, woff2b, 128,  64, nwoff2);
    k_prep_w<<<blk(nw2),    256, 0, stream>>>(w2,     w2b,     64,  64, nw2);
    k_x_to_nd<<<blk(NIMG * 256 * HW), 256, 0, stream>>>(x, xnd, NIMG * 256 * HW);

    // stage 1: offset conv (256 -> 512), sample, conv1 (256 -> 64) + BN/LReLU
    k_conv3x3_wmma<<<dim3(16, 8, NIMG), 256, 0, stream>>>(
        xnd, woff1b, 256, offbuf, nullptr, nullptr, nullptr, nullptr, nullptr, 0);
    k_deform_sample<<<blk(NIMG * 256 * HW), 256, 0, stream>>>(
        x, 1, offbuf, 256, sampb, NIMG * 256 * HW);
    k_conv3x3_wmma<<<dim3(16, 1, NIMG), 256, 0, stream>>>(
        sampb, w1b, 256, midf, midb, g1, b1, m1, v1, 1);

    // stage 2: offset conv (64 -> 128), sample, conv2 (64 -> 64) + BN/LReLU,
    //          store transposed into d_out (B,64,T,H,W)
    k_conv3x3_wmma<<<dim3(16, 2, NIMG), 256, 0, stream>>>(
        midb, woff2b, 64, offbuf, nullptr, nullptr, nullptr, nullptr, nullptr, 0);
    k_deform_sample<<<blk(NIMG * 64 * HW), 256, 0, stream>>>(
        midf, 0, offbuf, 64, sampb, NIMG * 64 * HW);
    k_conv3x3_wmma<<<dim3(16, 1, NIMG), 256, 0, stream>>>(
        sampb, w2b, 64, (float*)d_out, nullptr, g2, b2, m2, v2, 2);
}
